// general_attention_49065706389913
// MI455X (gfx1250) — matrix operand order repaired and checked by reading, not compiled, not run on hardware
//
#include <hip/hip_runtime.h>

typedef __attribute__((ext_vector_type(16))) _Float16 v16h;
typedef __attribute__((ext_vector_type(8)))  _Float16 v8h;
typedef __attribute__((ext_vector_type(8)))  float    v8f;

static __device__ __forceinline__ v8f wmma_f16(v16h a, v16h b, v8f c) {
  // D = A(16x32,f16) * B(32x16,f16) + C(16x16,f32)
  return __builtin_amdgcn_wmma_f32_16x16x32_f16(
      /*neg_a=*/false, a, /*neg_b=*/false, b,
      /*c_mod=*/(short)0, c, /*reuse_a=*/false, /*reuse_b=*/false);
}

#define BATCH 4
#define HEADS 16
#define SEQ   2048
#define DIM   64
#define QTILES (SEQ / 16)   // 128
#define SR    2064          // f16 LDS row stride (halves): 16B-aligned rows, bank offset 8/row
#define MASKVAL (-60000.0f) // f16-representable; exp() underflows to 0 like -1e9

// ---------------------------------------------------------------------------
// Kernel 1: k_proj = K @ W^T (f16 out, f32 WMMA accumulate) + V transpose
//   kp16[bh][k][d] (d contiguous)    v16T[bh][e][k] (k contiguous)
// grid: (B*H)*128 key-tiles, 32 threads (1 wave)
// ---------------------------------------------------------------------------
__global__ void kproj_vtrans_kernel(const float* __restrict__ key,
                                    const float* __restrict__ value,
                                    const float* __restrict__ W,
                                    _Float16* __restrict__ kp16,
                                    _Float16* __restrict__ v16T) {
  const int bh   = blockIdx.x >> 7;
  const int kt   = blockIdx.x & 127;
  const int lane = threadIdx.x;
  const int hi   = (lane >= 16);
  const int m    = lane & 15;

  const float* krow  = key + ((size_t)(bh * SEQ + kt * 16 + m)) * DIM;
  const int    abase = hi ? 16 : 0;
  v16h a0, a1;
#pragma unroll
  for (int i = 0; i < 8; ++i) {
    a0[i]     = (_Float16)krow[abase + i];
    a0[8 + i] = (_Float16)krow[8 + abase + i];
    a1[i]     = (_Float16)krow[32 + abase + i];
    a1[8 + i] = (_Float16)krow[40 + abase + i];
  }

  const int dbase = hi ? 16 : 0;
#pragma unroll
  for (int et = 0; et < 4; ++et) {
    const int    e    = et * 16 + m;
    const float* wrow = W + e * DIM;              // B[d,e] = W[e,d], contiguous in d
    v16h b0, b1;
#pragma unroll
    for (int j = 0; j < 16; ++j) {
      b0[j] = (_Float16)wrow[dbase + j];
      b1[j] = (_Float16)wrow[32 + dbase + j];
    }
    v8f c = {};
    c = wmma_f16(a0, b0, c);
    c = wmma_f16(a1, b1, c);
#pragma unroll
    for (int r = 0; r < 8; ++r) {
      const int mm = r + (hi ? 8 : 0);
      kp16[((size_t)(bh * SEQ + kt * 16 + mm)) * DIM + e] = (_Float16)c[r];
    }
  }

  for (int idx = lane; idx < 16 * DIM; idx += 32) {
    const int   kl = idx >> 6;
    const int   e  = idx & 63;
    const float v  = value[((size_t)(bh * SEQ + kt * 16 + kl)) * DIM + e];
    v16T[((size_t)bh * DIM + e) * SEQ + kt * 16 + kl] = (_Float16)v;
  }
}

// ---------------------------------------------------------------------------
// Kernel 2: per (bh, 16-query tile): scores -> mask -> softmax -> prob, P@V
// grid: (B*H)*128 q-tiles, 128 threads (4 waves), ~65KB LDS -> 4 WG/WGP
// ---------------------------------------------------------------------------
__global__ void attention_kernel(const float* __restrict__ query,
                                 const unsigned char* __restrict__ mask,
                                 const _Float16* __restrict__ kp16,
                                 const _Float16* __restrict__ v16T,
                                 float* __restrict__ out,
                                 float* __restrict__ prob) {
  extern __shared__ char smem_raw[];
  _Float16* sE     = (_Float16*)smem_raw;        // 16 * SR halves: scores, then exp
  float*    red    = (float*)(sE + 16 * SR);     // 128 partials
  float*    rowmax = red + 128;                  // 16
  float*    rowinv = rowmax + 16;                // 16

  const int bh   = blockIdx.x >> 7;
  const int qt   = blockIdx.x & 127;
  const int b    = bh >> 4;                      // H == 16
  const int tid  = threadIdx.x;
  const int w    = tid >> 5;
  const int lane = tid & 31;
  const int hi   = (lane >= 16);
  const int m    = lane & 15;

  // ---- Phase 1: raw scores = (Q @ kp^T)/8 -> f16 LDS (mask applied later) ----
  const int    abase = hi ? 16 : 0;
  const int    dbase = hi ? 16 : 0;
  const float* qrow  = query + ((size_t)(bh * SEQ + qt * 16 + m)) * DIM;
  v16h qa0, qa1;
#pragma unroll
  for (int i = 0; i < 8; ++i) {
    qa0[i]     = (_Float16)qrow[abase + i];
    qa0[8 + i] = (_Float16)qrow[8 + abase + i];
    qa1[i]     = (_Float16)qrow[32 + abase + i];
    qa1[8 + i] = (_Float16)qrow[40 + abase + i];
  }

  for (int kt = w; kt < QTILES; kt += 4) {
    const int       kg    = kt * 16 + m;
    const _Float16* kprow = kp16 + ((size_t)(bh * SEQ + kg)) * DIM;
    v16h b0, b1;
#pragma unroll
    for (int j = 0; j < 16; ++j) {               // 2 contiguous 32B loads
      b0[j] = kprow[dbase + j];
      b1[j] = kprow[32 + dbase + j];
    }
    v8f c = {};
    c = wmma_f16(qa0, b0, c);
    c = wmma_f16(qa1, b1, c);
#pragma unroll
    for (int r = 0; r < 8; ++r) {
      const int mm = r + (hi ? 8 : 0);
      sE[mm * SR + kg] = (_Float16)(c[r] * 0.125f);  // 1/sqrt(64)
    }
  }
  __syncthreads();

  // ---- Phase 2: apply mask (coalesced 8B loads) + row max; thread = (row, 256-chunk) ----
  const int r2 = tid & 15;
  const int c2 = tid >> 4;
  _Float16* rowp = sE + r2 * SR + c2 * 256;
  const unsigned char* mrow =
      mask + ((size_t)b * SEQ + (qt * 16 + r2)) * SEQ + c2 * 256;
  float mx = -3.4e38f;
  for (int j = 0; j < 32; ++j) {
    v8h s8 = *(const v8h*)(rowp + j * 8);
    const unsigned long long mb = *(const unsigned long long*)(mrow + j * 8);
#pragma unroll
    for (int i = 0; i < 8; ++i) {
      float s = (float)s8[i];
      if ((mb >> (8 * i)) & 0xffull) s = MASKVAL;
      s8[i] = (_Float16)s;
      mx = fmaxf(mx, s);
    }
    *(v8h*)(rowp + j * 8) = s8;
  }
  red[tid] = mx;                                 // red[c2*16 + r2]
  __syncthreads();
  if (tid < 16) {
    float v = red[tid];
#pragma unroll
    for (int c = 1; c < 8; ++c) v = fmaxf(v, red[tid + 16 * c]);
    rowmax[tid] = v;
  }
  __syncthreads();

  // ---- Phase 3: exp (stored back as f16, unnormalized) + row sum -> 1/sum ----
  const float rm  = rowmax[r2];
  float       sum = 0.f;
  for (int j = 0; j < 32; ++j) {
    v8h s8 = *(const v8h*)(rowp + j * 8);
#pragma unroll
    for (int i = 0; i < 8; ++i) {
      const float e = __expf((float)s8[i] - rm);
      s8[i] = (_Float16)e;
      sum += e;
    }
    *(v8h*)(rowp + j * 8) = s8;
  }
  red[tid] = sum;
  __syncthreads();
  if (tid < 16) {
    float s2 = 0.f;
#pragma unroll
    for (int c = 0; c < 8; ++c) s2 += red[tid + 16 * c];
    rowinv[tid] = 1.0f / s2;
  }
  __syncthreads();

  // ---- Phase 4: normalized prob, coalesced float4 global writes (the 1GB output) ----
  {
    float* pr = prob + ((size_t)(bh * SEQ + qt * 16)) * SEQ;
#pragma unroll
    for (int rr = 0; rr < 16; ++rr) {
      const float     iv  = rowinv[rr];
      const _Float16* src = sE + rr * SR;
      float*          dst = pr + (size_t)rr * SEQ;
#pragma unroll
      for (int i = 0; i < 4; ++i) {
        const int       col = tid * 4 + i * 512;
        const _Float16* sp  = src + col;         // 8B-aligned -> ds_load_b64
        float4 v4;
        v4.x = (float)sp[0] * iv;
        v4.y = (float)sp[1] * iv;
        v4.z = (float)sp[2] * iv;
        v4.w = (float)sp[3] * iv;
        *(float4*)(dst + col) = v4;
      }
    }
  }

  // ---- Phase 5: out = diag(1/sum) * (E @ V); wave w owns e-tile w; A = raw exp f16 ----
  v8f             acc  = {};
  const _Float16* prow = sE + m * SR;            // A row M = lane%16
  const int       e    = w * 16 + m;             // B col N
  const _Float16* vcol = v16T + ((size_t)bh * DIM + e) * SEQ;
  for (int k0 = 0; k0 < SEQ; k0 += 32) {
    v16h pa;
    const _Float16* p0 = prow + k0 + abase;      // two 16B ds_load_b128, zero VALU
#pragma unroll
    for (int i = 0; i < 8; ++i) {
      pa[i]     = p0[i];
      pa[8 + i] = p0[8 + i];
    }
    v16h vb;
    const _Float16* vp = vcol + k0 + dbase;      // contiguous 32B global
#pragma unroll
    for (int j = 0; j < 16; ++j) vb[j] = vp[j];
    acc = wmma_f16(pa, vb, acc);
  }
#pragma unroll
  for (int r = 0; r < 8; ++r) {
    const int mm = r + (hi ? 8 : 0);             // C/D row
    out[((size_t)(bh * SEQ + qt * 16 + mm)) * DIM + e] = acc[r] * rowinv[mm];
  }
}

extern "C" void kernel_launch(void* const* d_in, const int* in_sizes, int n_in,
                              void* d_out, int out_size, void* d_ws, size_t ws_size,
                              hipStream_t stream) {
  const float*         q    = (const float*)d_in[0];
  const float*         k    = (const float*)d_in[1];
  const float*         v    = (const float*)d_in[2];
  const float*         W    = (const float*)d_in[3];
  const unsigned char* mask = (const unsigned char*)d_in[4];
  // d_in[5] = dropout scalar (0), unused

  float* out  = (float*)d_out;
  float* prob = out + (size_t)BATCH * HEADS * SEQ * DIM;      // tuple: (out, prob_attn)

  _Float16* kp16 = (_Float16*)d_ws;                           // 16 MB
  _Float16* v16T = kp16 + (size_t)BATCH * HEADS * SEQ * DIM;  // 16 MB

  const int nbh = BATCH * HEADS;                              // 64

  kproj_vtrans_kernel<<<nbh * QTILES, 32, 0, stream>>>(k, v, W, kp16, v16T);

  const size_t shmem = (size_t)16 * SR * sizeof(_Float16) +
                       (size_t)(128 + 16 + 16) * sizeof(float); // ~65.8 KB
  (void)hipFuncSetAttribute((const void*)attention_kernel,
                            hipFuncAttributeMaxDynamicSharedMemorySize, (int)shmem);
  attention_kernel<<<nbh * QTILES, 128, shmem, stream>>>(q, mask, kp16, v16T, out, prob);
}
